// GCNLayer_1090921693296
// MI455X (gfx1250) — matrix operand order repaired and checked by reading, not compiled, not run on hardware
//
#include <hip/hip_runtime.h>

typedef __attribute__((ext_vector_type(16))) __bf16 v16bf;
typedef __attribute__((ext_vector_type(8)))  __bf16 v8bf;
typedef __attribute__((ext_vector_type(8)))  float  v8f;

#define D_DIM 512

// ---------------------------------------------------------------- zero output
__global__ void zero_f4(float4* __restrict__ p, int n4) {
  int i = blockIdx.x * blockDim.x + threadIdx.x;
  int stride = gridDim.x * blockDim.x;
  float4 z = make_float4(0.f, 0.f, 0.f, 0.f);
  for (; i < n4; i += stride) p[i] = z;
}

// -------------------------------- W: f32 row-major [K][N] -> bf16 Wt[N][K]
// idx = k*512 + n (n fastest) -> coalesced reads; scattered 2B writes (tiny).
__global__ void wtrans(const float* __restrict__ W, __bf16* __restrict__ Wt) {
  int idx = blockIdx.x * blockDim.x + threadIdx.x;   // 0 .. 512*512-1
  int k = idx >> 9;
  int n = idx & 511;
  Wt[(size_t)n * D_DIM + k] = (__bf16)W[idx];
}

// ---------------------------------------------------------------- GEMM (WMMA)
// Block = 256 threads = 8 waves. Block computes a 16 x 512 slab of `support`.
// Wave w computes cols [64w, 64w+64) as 4 accumulator tiles of 16x16.
__global__ __launch_bounds__(256) void gemm_bf16(
    const float* __restrict__ x, const __bf16* __restrict__ Wt,
    float* __restrict__ support, int Nrows)
{
  // 520-element rows: 1040B row stride -> 16B aligned, bank-conflict free.
  __shared__ __align__(16) __bf16 sA[16][520];
  const int row0 = blockIdx.x * 16;

  // Cooperative stage: 16x512 f32 -> bf16 in LDS (each thread 32 floats).
  {
    int t = threadIdx.x;
    int r = t >> 4;              // 16 threads per row
    int c = (t & 15) * 32;
    bool valid = (row0 + r) < Nrows;
    const float* src = x + (size_t)(row0 + r) * D_DIM + c;
#pragma unroll
    for (int i = 0; i < 32; i += 4) {
      float4 v = valid ? *(const float4*)(src + i) : make_float4(0.f, 0.f, 0.f, 0.f);
      sA[r][c + i + 0] = (__bf16)v.x;
      sA[r][c + i + 1] = (__bf16)v.y;
      sA[r][c + i + 2] = (__bf16)v.z;
      sA[r][c + i + 3] = (__bf16)v.w;
    }
  }
  __syncthreads();

  const int wave = threadIdx.x >> 5;
  const int lane = threadIdx.x & 31;
  const int hi   = lane >> 4;       // lane group (0: lanes 0-15, 1: 16-31)
  const int mn   = lane & 15;       // row (A) / col (B,C,D) index
  const int c0   = wave * 64;

  v8f acc0 = {}, acc1 = {}, acc2 = {}, acc3 = {};

  for (int k0 = 0; k0 < D_DIM; k0 += 32) {
    // A fragment (contiguous k order, matches B):
    //   elems 0-7  = A[mn][k0 + hi*16 + j]
    //   elems 8-15 = A[mn][k0 + hi*16 + 8 + j]
    v8bf alo = *(const v8bf*)&sA[mn][k0 + hi * 16];
    v8bf ahi = *(const v8bf*)&sA[mn][k0 + hi * 16 + 8];
    v16bf a;
#pragma unroll
    for (int j = 0; j < 8; ++j) { a[j] = alo[j]; a[8 + j] = ahi[j]; }

    // B fragment, contiguous k order (same as A): lane holds col n = mn,
    // elems j = B[k0 + hi*16 + j][n]  -> 32 contiguous bytes of Wt[n][*].
    const __bf16* bbase = Wt + (size_t)(c0 + mn) * D_DIM + k0 + hi * 16;
    v16bf b0 = *(const v16bf*)(bbase + (size_t)0 * 16 * D_DIM);
    v16bf b1 = *(const v16bf*)(bbase + (size_t)1 * 16 * D_DIM);
    v16bf b2 = *(const v16bf*)(bbase + (size_t)2 * 16 * D_DIM);
    v16bf b3 = *(const v16bf*)(bbase + (size_t)3 * 16 * D_DIM);

    acc0 = __builtin_amdgcn_wmma_f32_16x16x32_bf16(false, a, false, b0, (short)0, acc0, false, false);
    acc1 = __builtin_amdgcn_wmma_f32_16x16x32_bf16(false, a, false, b1, (short)0, acc1, false, false);
    acc2 = __builtin_amdgcn_wmma_f32_16x16x32_bf16(false, a, false, b2, (short)0, acc2, false, false);
    acc3 = __builtin_amdgcn_wmma_f32_16x16x32_bf16(false, a, false, b3, (short)0, acc3, false, false);
  }

  // C/D layout: elem v of lane = C[hi*8 + v][mn]
#pragma unroll
  for (int r = 0; r < 8; ++r) {
    int row = row0 + hi * 8 + r;
    if (row < Nrows) {
      float* o = support + (size_t)row * D_DIM + c0 + mn;
      o[0]  = acc0[r];
      o[16] = acc1[r];
      o[32] = acc2[r];
      o[48] = acc3[r];
    }
  }
}

// ------------------------------------------------- COO scatter (SpMM phase)
// One wave per edge: gather 512 f32 from support[src], scale, atomic-add
// into out[dst]. Working set (support + out = ~205 MB... both ~102 MB) is
// L2-resident (192 MB holds the hot out rows + support), atomics hit L2.
__global__ __launch_bounds__(256) void spmm_scatter(
    const float* __restrict__ support, const float* __restrict__ ew,
    const int* __restrict__ esrc, const int* __restrict__ edst,
    float* __restrict__ out, int E)
{
  int e = (int)((blockIdx.x * (unsigned)blockDim.x + threadIdx.x) >> 5);
  if (e >= E) return;
  int lane = threadIdx.x & 31;
  int s = esrc[e];
  int d = edst[e];
  float w = ew[e];
  const float4* sp = (const float4*)(support + (size_t)s * D_DIM);
  float* op = out + (size_t)d * D_DIM;
#pragma unroll
  for (int i = 0; i < 4; ++i) {
    int idx = lane + 32 * i;
    float4 v = sp[idx];
    int c = idx * 4;
    unsafeAtomicAdd(op + c + 0, v.x * w);
    unsafeAtomicAdd(op + c + 1, v.y * w);
    unsafeAtomicAdd(op + c + 2, v.z * w);
    unsafeAtomicAdd(op + c + 3, v.w * w);
  }
}

// ---------------------------------------------------------------------------
extern "C" void kernel_launch(void* const* d_in, const int* in_sizes, int n_in,
                              void* d_out, int out_size, void* d_ws, size_t ws_size,
                              hipStream_t stream) {
  const float* x    = (const float*)d_in[0];
  const float* W    = (const float*)d_in[1];
  const float* ew   = (const float*)d_in[2];
  const int*   esrc = (const int*)d_in[3];
  const int*   edst = (const int*)d_in[4];
  float* out = (float*)d_out;

  const int D = D_DIM;
  const int N = in_sizes[0] / D;   // 50000
  const int E = in_sizes[2];       // 800000

  // Workspace: [support f32 N*D][Wt bf16 D*D]
  char* ws = (char*)d_ws;
  float* support = (float*)ws;
  size_t supBytes = (((size_t)N * D * sizeof(float)) + 255) & ~(size_t)255;
  __bf16* Wt = (__bf16*)(ws + supBytes);

  // 1) zero output (scatter accumulates into it)
  int n4 = out_size / 4;
  zero_f4<<<4096, 256, 0, stream>>>((float4*)out, n4);

  // 2) convert + transpose W -> bf16 Wt[N][K]
  wtrans<<<(D * D) / 256, 256, 0, stream>>>(W, Wt);

  // 3) support = x @ W via v_wmma_f32_16x16x32_bf16
  gemm_bf16<<<(N + 15) / 16, 256, 0, stream>>>(x, Wt, support, N);

  // 4) out[dst] += ew * support[src]
  spmm_scatter<<<(E + 7) / 8, 256, 0, stream>>>(support, ew, esrc, edst, out, E);
}
